// CausalLiquidNetworkModel_30528627540773
// MI455X (gfx1250) — matrix operand order repaired and checked by reading, not compiled, not run on hardware
//
#include <hip/hip_runtime.h>
#include <hip/hip_bf16.h>

typedef _Float16 v16h __attribute__((ext_vector_type(16)));
typedef _Float16 h8   __attribute__((ext_vector_type(8)));
typedef float    v8f  __attribute__((ext_vector_type(8)));
typedef float    f4   __attribute__((ext_vector_type(4)));

#define LOG2E 1.4426950408889634f

__device__ __forceinline__ float fexp(float x)     { return __builtin_amdgcn_exp2f(x * LOG2E); }
__device__ __forceinline__ float fsigmoid(float x) { return __builtin_amdgcn_rcpf(1.0f + fexp(-x)); }
__device__ __forceinline__ float ftanh(float x)    { return 2.0f * __builtin_amdgcn_rcpf(1.0f + fexp(-2.0f * x)) - 1.0f; }

constexpr int   BB = 256;   // batch
constexpr int   SS = 2048;  // sequence
constexpr int   HH = 128;   // hidden
constexpr float DT_C = 0.1f;

// One workgroup = 16 batch rows. 8 waves; wave w owns hidden columns [w*16, w*16+16).
// h state lives in LDS as an f16 16x128 row-major tile (WMMA A-operand source).
// Weight B-operands (W^T fragments) are packed into registers once (loop-invariant).
// The A-fragments of h loaded for phase C are kept in registers and reused by the
// next step's phase A (same data), so phase A issues its WMMAs with no LDS loads.
__global__ __launch_bounds__(256)
void liquid_scan_kernel(const float* __restrict__ X,
                        const float* __restrict__ W_in,  const float* __restrict__ b_in,
                        const float* __restrict__ W_rec, const float* __restrict__ b_rec,
                        const float* __restrict__ tau,
                        const float* __restrict__ W_att, const float* __restrict__ b_att,
                        const float* __restrict__ W_ev,  const float* __restrict__ b_ev,
                        const float* __restrict__ W_acc, const float* __restrict__ b_acc,
                        const float* __restrict__ W_out, const float* __restrict__ b_out,
                        float* __restrict__ out, float* __restrict__ ewOut)
{
    __shared__ __align__(16) _Float16 hA[16 * HH];   // h (A-matrix layout source)
    __shared__ __align__(16) _Float16 tA[16 * HH];   // h * att (A-matrix layout source)
    __shared__ __align__(16) float xsh[2][16];       // x_t broadcast (double-buffered)
    __shared__ __align__(16) float ews[2][16];       // event weight broadcast (double-buffered)
    __shared__ float red[16];                        // output reduction

    const int tid = threadIdx.x;
    const int w   = tid >> 5;        // wave id 0..7
    const int ln  = tid & 31;        // lane in wave
    const int nl  = ln & 15;         // column (N) within 16-wide tile; also A-row M
    const int kh  = ln >> 4;         // lane-half: selects K-half (A/B) and M-half (C/D)
    const int kh8 = kh * 8;
    const int n   = w * 16 + nl;     // global hidden index owned by this lane
    const int bt  = blockIdx.x;      // batch tile (16 rows)

    // Per-lane loop-invariant scalars (hidden index n fixed per lane).
    const float batt = b_att[n], brec = b_rec[n], bacc = b_acc[n];
    const float win  = W_in[n],  bin  = b_in[n],  wout = W_out[n];
    float tc = tau[n]; tc = fminf(fmaxf(tc, 0.1f), 10.0f);
    const float itau = 1.0f / tc;

    // B-operand fragments: B[k][n] = W[n][k]. Lane (N=n) holds K = kf*32 + kh*16 .. +15,
    // packed 2 f16 per VGPR (contiguous k order, same as the A loads: lanes 0-15 K lo-half, 16-31 hi).
    v16h wAtt[4], wRec[4], wAcc[4];
    #pragma unroll
    for (int kf = 0; kf < 4; ++kf) {
        const float* pa = W_att + n * HH + kf * 32 + kh * 16;
        const float* pr = W_rec + n * HH + kf * 32 + kh * 16;
        const float* pc = W_acc + n * HH + kf * 32 + kh * 16;
        #pragma unroll
        for (int i = 0; i < 16; ++i) {
            wAtt[kf][i] = (_Float16)pa[i];
            wRec[kf][i] = (_Float16)pr[i];
            wAcc[kf][i] = (_Float16)pc[i];
        }
    }

    // h0 = 0 in LDS.
    for (int i = tid; i < 16 * HH; i += 256) hA[i] = (_Float16)0.0f;

    v8f h = {}, acc = {};
    v16h aF[4] = {};                 // A fragments of current h (h0 = 0)
    const int aOff = nl * HH;        // A-operand row base (M = nl)

    // Wave-0 lanes 0..15 handle x loads / event weights for the 16 batch rows.
    const bool w0l = (tid < 16);
    const int  row = bt * 16 + (tid & 15);
    float wev0 = 0.f, wev1 = 0.f, bev = 0.f, xcur = 0.f, xprev = 0.f;
    if (w0l) {
        wev0 = W_ev[0]; wev1 = W_ev[1]; bev = b_ev[0];
        xcur = X[(size_t)row * SS];  // prefetch x at t=0
    }

    __syncthreads();

    #pragma clang loop unroll(disable)
    for (int t = 0; t < SS; ++t) {
        const int pb = t & 1;

        // ---------------- phase A: att = sigmoid(h @ W_att^T + b_att) ----------------
        // (aF already holds the A fragments of h — no LDS loads here.)
        if (w0l) {
            float e = 0.0f;
            if (t > 0) e = fsigmoid(wev0 * xcur + wev1 * xprev + bev);
            ews[pb][tid] = e;
            xsh[pb][tid] = xcur;
            ewOut[(size_t)row * SS + t] = e;
            float xn = 0.0f;
            if (t + 1 < SS) xn = X[(size_t)row * SS + t + 1];  // prefetch next step
            xprev = xcur; xcur = xn;
        }

        v8f am = {};
        #pragma unroll
        for (int kf = 0; kf < 4; ++kf)
            am = __builtin_amdgcn_wmma_f32_16x16x32_f16(false, aF[kf], false, wAtt[kf],
                                                        (short)0, am, false, false);
        // h * att -> tA (f16, A layout). D-tile row M = kh8 + r, col = n.
        #pragma unroll
        for (int r = 0; r < 8; ++r) {
            float att = fsigmoid(am[r] + batt);
            tA[(kh8 + r) * HH + n] = (_Float16)(h[r] * att);
        }
        __syncthreads();

        // ---------------- phase B: rec, dh, h_new ----------------
        v16h bF[4];
        #pragma unroll
        for (int kf = 0; kf < 4; ++kf) {
            h8 lo = *(const h8*)(&tA[aOff + kf * 32 + kh8 * 2]);
            h8 hi = *(const h8*)(&tA[aOff + kf * 32 + kh8 * 2 + 8]);
            bF[kf] = __builtin_shufflevector(lo, hi, 0,1,2,3,4,5,6,7,8,9,10,11,12,13,14,15);
        }
        v8f rm = {};
        #pragma unroll
        for (int kf = 0; kf < 4; ++kf)
            rm = __builtin_amdgcn_wmma_f32_16x16x32_f16(false, bF[kf], false, wRec[kf],
                                                        (short)0, rm, false, false);
        {
            f4 xv0 = *(const f4*)(&xsh[pb][kh8]);
            f4 xv1 = *(const f4*)(&xsh[pb][kh8 + 4]);
            f4 ev0 = *(const f4*)(&ews[pb][kh8]);
            f4 ev1 = *(const f4*)(&ews[pb][kh8 + 4]);
            #pragma unroll
            for (int r = 0; r < 8; ++r) {
                float xv  = (r < 4) ? xv0[r & 3] : xv1[r & 3];
                float e   = (r < 4) ? ev0[r & 3] : ev1[r & 3];
                float ic  = ftanh(xv * win + bin);
                float rec = ftanh(rm[r] + brec);
                float dh  = (-h[r] + ic + rec) * itau;
                h[r] = h[r] + DT_C * dh * (1.0f + e);
                hA[(kh8 + r) * HH + n] = (_Float16)h[r];   // h_new -> LDS
            }
        }
        __syncthreads();

        // ---- phase C: acc = 0.9 acc + 0.1 tanh(h_new @ W_acc^T + b_acc) * ew ----
        // Load h_new fragments into aF (also serves next step's phase A).
        #pragma unroll
        for (int kf = 0; kf < 4; ++kf) {
            h8 lo = *(const h8*)(&hA[aOff + kf * 32 + kh8 * 2]);
            h8 hi = *(const h8*)(&hA[aOff + kf * 32 + kh8 * 2 + 8]);
            aF[kf] = __builtin_shufflevector(lo, hi, 0,1,2,3,4,5,6,7,8,9,10,11,12,13,14,15);
        }
        v8f cm = {};
        #pragma unroll
        for (int kf = 0; kf < 4; ++kf)
            cm = __builtin_amdgcn_wmma_f32_16x16x32_f16(false, aF[kf], false, wAcc[kf],
                                                        (short)0, cm, false, false);
        {
            f4 ev0 = *(const f4*)(&ews[pb][kh8]);
            f4 ev1 = *(const f4*)(&ews[pb][kh8 + 4]);
            #pragma unroll
            for (int r = 0; r < 8; ++r) {
                float e = (r < 4) ? ev0[r & 3] : ev1[r & 3];
                acc[r] = 0.9f * acc[r] + 0.1f * ftanh(cm[r] + bacc) * e;
            }
        }
        // No barrier needed here: next phase A only reads aF (registers), writes tA
        // and the opposite-parity ews/xsh buffers; bar1 of the next step orders hA.
    }

    // ---------------- output: (h + acc) @ W_out^T + b_out ----------------
    __syncthreads();
    if (tid < 16) red[tid] = 0.0f;
    __syncthreads();
    #pragma unroll
    for (int r = 0; r < 8; ++r)
        atomicAdd(&red[kh8 + r], (h[r] + acc[r]) * wout);
    __syncthreads();
    if (tid < 16) out[bt * 16 + tid] = red[tid] + b_out[0];
}

extern "C" void kernel_launch(void* const* d_in, const int* in_sizes, int n_in,
                              void* d_out, int out_size, void* d_ws, size_t ws_size,
                              hipStream_t stream) {
    (void)in_sizes; (void)n_in; (void)d_ws; (void)ws_size; (void)out_size;
    const float* X     = (const float*)d_in[0];
    const float* W_in  = (const float*)d_in[1];
    const float* b_in  = (const float*)d_in[2];
    const float* W_rec = (const float*)d_in[3];
    const float* b_rec = (const float*)d_in[4];
    const float* tau   = (const float*)d_in[5];
    const float* W_att = (const float*)d_in[6];
    const float* b_att = (const float*)d_in[7];
    const float* W_ev  = (const float*)d_in[8];
    const float* b_ev  = (const float*)d_in[9];
    const float* W_acc = (const float*)d_in[10];
    const float* b_acc = (const float*)d_in[11];
    const float* W_out = (const float*)d_in[12];
    const float* b_out = (const float*)d_in[13];

    float* out = (float*)d_out;        // [B, O] = 256 floats
    float* ew  = out + BB;             // [B, S] = 256*2048 floats

    liquid_scan_kernel<<<BB / 16, 256, 0, stream>>>(
        X, W_in, b_in, W_rec, b_rec, tau, W_att, b_att,
        W_ev, b_ev, W_acc, b_acc, W_out, b_out, out, ew);
}
